// MultiHeadAttention_81930796138862
// MI455X (gfx1250) — matrix operand order repaired and checked by reading, not compiled, not run on hardware
//
#include <hip/hip_runtime.h>
#include <hip/hip_bf16.h>

// ---------------------------------------------------------------------------
// MoE attention pipeline for MI455X (gfx1250, wave32, WMMA bf16 + TDM).
// ---------------------------------------------------------------------------

typedef __attribute__((ext_vector_type(16))) __bf16       v16bf;
typedef __attribute__((ext_vector_type(8)))  float        v8f;
typedef __attribute__((ext_vector_type(4)))  unsigned int u32x4;
typedef __attribute__((ext_vector_type(4)))  int          i32x4;
typedef __attribute__((ext_vector_type(8)))  int          i32x8;

#define N_TOK   8192      // B*T
#define B_SZ    4
#define T_SEQ   2048
#define D_MODEL 512
#define D_KEY   64
#define N_EXP   8
#define TOP_K   2

// Tensor Data Mover availability (device pass only; falls back cleanly).
#if defined(__HIP_DEVICE_COMPILE__) && \
    __has_builtin(__builtin_amdgcn_tensor_load_to_lds) && \
    __has_builtin(__builtin_amdgcn_s_wait_tensorcnt)
#define HAS_TDM 1
#else
#define HAS_TDM 0
#endif

template <bool RA>
static __device__ __forceinline__ v8f wmma_bf16_t(v16bf a, v16bf b, v8f c) {
    return __builtin_amdgcn_wmma_f32_16x16x32_bf16(
        /*neg_a=*/false, a, /*neg_b=*/false, b,
        /*c_mod=*/(short)0, c, /*reuse_a=*/RA, /*reuse_b=*/false);
}

static __device__ __forceinline__ void v8f_zero(v8f& v) {
    #pragma unroll
    for (int i = 0; i < 8; ++i) v[i] = 0.0f;
}

// A-fragment element index: 16-bit A 16x32, half h holds K = h*16+j;
// row M = lane&15.
static __device__ __forceinline__ int a_kidx(int h, int j) {
    return h * 16 + j;
}

// B-fragment gather from an LDS tile with leading dim ldm: element j is
// B[h*16 + j][nt*16 + ln].
static __device__ __forceinline__ v16bf load_bfrag(const __bf16* base, int ldm,
                                                   int nt, int h, int ln) {
    v16bf b;
    #pragma unroll
    for (int j = 0; j < 16; ++j) b[j] = base[(h * 16 + j) * ldm + nt * 16 + ln];
    return b;
}

#if HAS_TDM
// Issue one TDM 2D tile load (bf16 elements) into LDS.
// D# group0: count=1 | lds_addr | global_addr | type=2 ("image").
// D# group1: data_size=2B, tensor_dim0/1, tile_dim0/1, dim0 stride.
static __device__ __forceinline__ void tdm_load_tile_bf16(
    void* lds_dst, const __bf16* gsrc,
    unsigned tensor_d0, unsigned tensor_d1, unsigned d0_stride,
    unsigned tile_d0, unsigned tile_d1)
{
    unsigned long long ga = (unsigned long long)(size_t)gsrc;
    unsigned lds_off = (unsigned)(unsigned long long)(size_t)lds_dst;
    u32x4 g0;
    g0[0] = 1u;                                        // count=1 (user mode)
    g0[1] = lds_off;                                   // lds_addr
    g0[2] = (unsigned)(ga & 0xFFFFFFFFu);              // global_addr[31:0]
    g0[3] = (unsigned)((ga >> 32) & 0x1FFFFFFu) | (2u << 30); // [56:32]|type=2
    i32x8 g1;
    g1[0] = (int)(1u << 16);                           // data_size = 2 bytes
    g1[1] = (int)((tensor_d0 & 0xFFFFu) << 16);        // tensor_dim0 lo
    g1[2] = (int)((tensor_d0 >> 16) | ((tensor_d1 & 0xFFFFu) << 16));
    g1[3] = (int)((tensor_d1 >> 16) | ((tile_d0 & 0xFFFFu) << 16));
    g1[4] = (int)(tile_d1 & 0xFFFFu);                  // tile_dim1 (tile_dim2=0)
    g1[5] = (int)d0_stride;                            // dim0 stride [31:0]
    g1[6] = 0;
    g1[7] = 0;
    i32x4 z4;
    z4[0] = 0; z4[1] = 0; z4[2] = 0; z4[3] = 0;
#if defined(__clang_major__) && (__clang_major__ >= 23)
    i32x8 z8;
    #pragma unroll
    for (int i = 0; i < 8; ++i) z8[i] = 0;
    __builtin_amdgcn_tensor_load_to_lds(g0, g1, z4, z4, z8, 0);
#else
    __builtin_amdgcn_tensor_load_to_lds(g0, g1, z4, z4, 0);
#endif
}
#endif // HAS_TDM

// ---------------------------------------------------------------------------
// fp32 -> bf16 bulk convert
// ---------------------------------------------------------------------------
__global__ void cvt_bf16_kernel(const float* __restrict__ src,
                                __bf16* __restrict__ dst, int n) {
    int i = blockIdx.x * blockDim.x + threadIdx.x;
    if (i < n) dst[i] = (__bf16)src[i];
}

// ---------------------------------------------------------------------------
// zero a dword region (dense expert-combine buffer c)
// ---------------------------------------------------------------------------
__global__ void zero_kernel(unsigned int* __restrict__ p, int n) {
    int i = blockIdx.x * blockDim.x + threadIdx.x;
    if (i < n) p[i] = 0u;
}

// ---------------------------------------------------------------------------
// Router: one wave per token. logits = q . Wg, softmax, top-2.
// ---------------------------------------------------------------------------
__global__ __launch_bounds__(256) void router_kernel(
    const float* __restrict__ q,     // [N, 512]
    const float* __restrict__ Wg,    // [512, 8]
    float* __restrict__ gates,       // [N, 2]
    int* __restrict__ idxs)          // [N, 2]
{
    int lane = threadIdx.x & 31;
    int tok  = blockIdx.x * (blockDim.x >> 5) + (threadIdx.x >> 5);
    if (tok >= N_TOK) return;
    const float* x = q + (size_t)tok * D_MODEL;

    float acc[N_EXP];
    #pragma unroll
    for (int e = 0; e < N_EXP; ++e) acc[e] = 0.0f;

    for (int i = lane; i < D_MODEL; i += 32) {
        float xv = x[i];
        const float* wr = Wg + i * N_EXP;
        #pragma unroll
        for (int e = 0; e < N_EXP; ++e) acc[e] += xv * wr[e];
    }
    #pragma unroll
    for (int e = 0; e < N_EXP; ++e)
        for (int off = 16; off; off >>= 1)
            acc[e] += __shfl_xor(acc[e], off, 32);

    float m = acc[0];
    #pragma unroll
    for (int e = 1; e < N_EXP; ++e) m = fmaxf(m, acc[e]);
    float p[N_EXP], s = 0.0f;
    #pragma unroll
    for (int e = 0; e < N_EXP; ++e) { p[e] = __expf(acc[e] - m); s += p[e]; }

    int i1 = 0; float v1 = p[0];
    #pragma unroll
    for (int e = 1; e < N_EXP; ++e) if (p[e] > v1) { v1 = p[e]; i1 = e; }
    int i2 = -1; float v2 = -1.0f;
    #pragma unroll
    for (int e = 0; e < N_EXP; ++e)
        if (e != i1 && p[e] > v2) { v2 = p[e]; i2 = e; }

    if (lane == 0) {
        float inv = 1.0f / s;
        gates[tok * 2 + 0] = v1 * inv;
        gates[tok * 2 + 1] = v2 * inv;
        idxs[tok * 2 + 0] = i1;
        idxs[tok * 2 + 1] = i2;
    }
}

// ---------------------------------------------------------------------------
// [16x512] x [512x64] projection tile (one wave). grid.y selects:
//   0 -> kh = key @ Wk + bk, 1 -> vh = value @ Wv + bv,
//   2+e -> q_all[e] = query @ Wq[e] + bq[e]
// Weight tile staged via TDM when available.
// ---------------------------------------------------------------------------
__global__ __launch_bounds__(32) void proj512x64_kernel(
    const float* __restrict__ Xq, const float* __restrict__ Xk,
    const float* __restrict__ Xv,
    const __bf16* __restrict__ WkB, const __bf16* __restrict__ WvB,
    const __bf16* __restrict__ WqB,              // [E][512][64]
    const float* __restrict__ bk, const float* __restrict__ bv,
    const float* __restrict__ bq,                // [E][64]
    __bf16* __restrict__ kh, __bf16* __restrict__ vh,
    __bf16* __restrict__ qa)                     // [E][N][64]
{
    __shared__ __attribute__((aligned(32))) __bf16 Ald[16 * 32];
    __shared__ __attribute__((aligned(32))) __bf16 Bld[32 * D_KEY];

    int lane = threadIdx.x & 31, h = lane >> 4, ln = lane & 15;
    int tile = blockIdx.x, sel = blockIdx.y;

    const float* X; const __bf16* W; const float* bias; __bf16* out;
    if (sel == 0)      { X = Xk; W = WkB; bias = bk; out = kh; }
    else if (sel == 1) { X = Xv; W = WvB; bias = bv; out = vh; }
    else {
        int e = sel - 2;
        X = Xq; W = WqB + (size_t)e * D_MODEL * D_KEY;
        bias = bq + e * D_KEY; out = qa + (size_t)e * N_TOK * D_KEY;
    }
    const float* xrow = X + (size_t)tile * 16 * D_MODEL;

    v8f acc[4];
    #pragma unroll
    for (int nt = 0; nt < 4; ++nt) v8f_zero(acc[nt]);

    for (int k0 = 0; k0 < D_MODEL; k0 += 32) {
        // stage B: 32 rows x 64 bf16 weight tile
#if HAS_TDM
        tdm_load_tile_bf16(Bld, W + (size_t)k0 * D_KEY,
                           /*tensor_d0=*/D_KEY, /*tensor_d1=*/D_MODEL - k0,
                           /*d0_stride=*/D_KEY, /*tile_d0=*/D_KEY, /*tile_d1=*/32);
#else
        {
            const uint4* s = (const uint4*)(W + (size_t)(k0 + lane) * D_KEY);
            uint4* d = (uint4*)(Bld + lane * D_KEY);
            #pragma unroll
            for (int qd = 0; qd < 8; ++qd) d[qd] = s[qd];
        }
#endif
        // stage A: 16 rows x 32 f32 -> bf16 (lane: row ln, 16 cols at h*16)
        {
            const float* s = xrow + (size_t)ln * D_MODEL + k0 + h * 16;
            __bf16* d = Ald + ln * 32 + h * 16;
            #pragma unroll
            for (int j = 0; j < 16; ++j) d[j] = (__bf16)s[j];
        }
#if HAS_TDM
        __builtin_amdgcn_s_wait_tensorcnt(0);
#endif
        __syncthreads();

        v16bf a;
        #pragma unroll
        for (int j = 0; j < 16; ++j) a[j] = Ald[ln * 32 + a_kidx(h, j)];
        // same A feeds 4 consecutive identical wmma -> reuse-A hint on 2..4
        acc[0] = wmma_bf16_t<false>(a, load_bfrag(Bld, D_KEY, 0, h, ln), acc[0]);
        acc[1] = wmma_bf16_t<true >(a, load_bfrag(Bld, D_KEY, 1, h, ln), acc[1]);
        acc[2] = wmma_bf16_t<true >(a, load_bfrag(Bld, D_KEY, 2, h, ln), acc[2]);
        acc[3] = wmma_bf16_t<true >(a, load_bfrag(Bld, D_KEY, 3, h, ln), acc[3]);
        __syncthreads();
    }

    #pragma unroll
    for (int nt = 0; nt < 4; ++nt) {
        float bb = bias[nt * 16 + ln];
        #pragma unroll
        for (int i = 0; i < 8; ++i) {
            int r = i + 8 * h;
            out[(size_t)(tile * 16 + r) * D_KEY + nt * 16 + ln] =
                (__bf16)(acc[nt][i] + bb);
        }
    }
}

// ---------------------------------------------------------------------------
// Gather top-k expert queries, fold in 1/sqrt(d) = 0.125 attention scale.
// ---------------------------------------------------------------------------
__global__ void gather_q_kernel(const __bf16* __restrict__ qa,  // [E][N][64]
                                const int* __restrict__ idxs,   // [N][2]
                                __bf16* __restrict__ qh)        // [B*2][T][64]
{
    int gid = blockIdx.x * blockDim.x + threadIdx.x;  // (n, k, 8-chunk)
    int chunk = gid & 7;
    int k = (gid >> 3) & 1;
    int n = gid >> 4;
    if (n >= N_TOK) return;
    int e = idxs[n * 2 + k];
    int b = n / T_SEQ, t = n % T_SEQ;
    const __bf16* s = qa + ((size_t)e * N_TOK + n) * D_KEY + chunk * 8;
    __bf16* d = qh + (((size_t)(b * TOP_K + k)) * T_SEQ + t) * D_KEY + chunk * 8;
    #pragma unroll
    for (int j = 0; j < 8; ++j) d[j] = (__bf16)((float)s[j] * 0.125f);
}

// ---------------------------------------------------------------------------
// Flash attention: one wave per (bh = b*2+kslot, 16-row Q tile).
// V chunks double-buffered through LDS via TDM (TENSORcnt pipelined).
// Scatters gate-weighted output into dense c[N][E*64].
// ---------------------------------------------------------------------------
__global__ __launch_bounds__(32) void attn_kernel(
    const __bf16* __restrict__ qh,    // [B*2][T][64] (pre-scaled)
    const __bf16* __restrict__ kh,    // [B][T][64]
    const __bf16* __restrict__ vh,    // [B][T][64]
    const float* __restrict__ gates,  // [N][2]
    const int* __restrict__ idxs,     // [N][2]
    __bf16* __restrict__ c)           // [N][512] (zeroed)
{
#if HAS_TDM
    constexpr int NBUF = 2;
#else
    constexpr int NBUF = 1;
#endif
    __shared__ __attribute__((aligned(32))) __bf16 Pld[16 * 32];
    __shared__ __attribute__((aligned(32))) __bf16 Vbuf[NBUF][32 * D_KEY];

    int lane = threadIdx.x & 31, h = lane >> 4, ln = lane & 15;
    int bh = blockIdx.y, b = bh >> 1, kslot = bh & 1;
    int t0 = blockIdx.x * 16;

    // Q A-fragments (rows t0..t0+15), K-dim d=64 -> two 16x32 fragments
    const __bf16* qbase = qh + ((size_t)bh * T_SEQ + t0) * D_KEY;
    v16bf aq[2];
    #pragma unroll
    for (int kb = 0; kb < 2; ++kb)
        #pragma unroll
        for (int j = 0; j < 16; ++j)
            aq[kb][j] = qbase[(size_t)ln * D_KEY + kb * 32 + a_kidx(h, j)];

    v8f o[4];
    #pragma unroll
    for (int nt = 0; nt < 4; ++nt) v8f_zero(o[nt]);
    float mrow[8], lrow[8];
    #pragma unroll
    for (int i = 0; i < 8; ++i) { mrow[i] = -1e30f; lrow[i] = 0.0f; }

    const __bf16* kbase = kh + (size_t)b * T_SEQ * D_KEY;
    const __bf16* vbase = vh + (size_t)b * T_SEQ * D_KEY;

#if HAS_TDM
    // prologue: async-load V chunk 0
    tdm_load_tile_bf16(&Vbuf[0][0], vbase,
                       D_KEY, T_SEQ, D_KEY, D_KEY, 32);
#endif

    for (int j0 = 0; j0 < T_SEQ; j0 += 32) {
        int buf = (j0 >> 5) % NBUF;
#if HAS_TDM
        if (j0 + 32 < T_SEQ) {
            tdm_load_tile_bf16(&Vbuf[buf ^ 1][0],
                               vbase + (size_t)(j0 + 32) * D_KEY,
                               D_KEY, T_SEQ - (j0 + 32), D_KEY, D_KEY, 32);
            __builtin_amdgcn_s_wait_tensorcnt(1);   // chunk j0 is resident
        } else {
            __builtin_amdgcn_s_wait_tensorcnt(0);
        }
#else
        if (j0 + 32 < T_SEQ) {
            __builtin_prefetch((const void*)(vbase + (size_t)(j0 + 32 + lane) * D_KEY), 0, 1);
        }
        {
            const uint4* s = (const uint4*)(vbase + (size_t)(j0 + lane) * D_KEY);
            uint4* d = (uint4*)(&Vbuf[0][0] + lane * D_KEY);
            #pragma unroll
            for (int qd = 0; qd < 8; ++qd) d[qd] = s[qd];
        }
#endif
        if (j0 + 32 < T_SEQ) {
            __builtin_prefetch((const void*)(kbase + (size_t)(j0 + 32 + lane) * D_KEY), 0, 1);
        }
        __syncthreads();
        const __bf16* Vld = &Vbuf[buf][0];

        // S = Q . K^T : 16 rows x 32 key-cols (two 16-col N tiles)
        v8f s0, s1; v8f_zero(s0); v8f_zero(s1);
        #pragma unroll
        for (int kb = 0; kb < 2; ++kb) {
            const __bf16* kr0 = kbase + (size_t)(j0 + ln) * D_KEY + kb * 32 + h * 16;
            const __bf16* kr1 = kbase + (size_t)(j0 + 16 + ln) * D_KEY + kb * 32 + h * 16;
            v16bf b0 = *(const v16bf*)kr0;
            v16bf b1 = *(const v16bf*)kr1;
            s0 = wmma_bf16_t<false>(aq[kb], b0, s0);
            s1 = wmma_bf16_t<true >(aq[kb], b1, s1);   // same A as previous
        }

        // online softmax over the 32-wide chunk
        #pragma unroll
        for (int i = 0; i < 8; ++i) {
            float v0 = s0[i], v1 = s1[i];
            float mx = fmaxf(v0, v1);
            for (int off = 8; off; off >>= 1)
                mx = fmaxf(mx, __shfl_xor(mx, off, 16));
            float mnew = fmaxf(mrow[i], mx);
            float sc = __expf(mrow[i] - mnew);
            float p0 = __expf(v0 - mnew);
            float p1 = __expf(v1 - mnew);
            float rs = p0 + p1;
            for (int off = 8; off; off >>= 1)
                rs += __shfl_xor(rs, off, 16);
            lrow[i] = lrow[i] * sc + rs;
            mrow[i] = mnew;
            #pragma unroll
            for (int nt = 0; nt < 4; ++nt) o[nt][i] *= sc;
            int r = i + 8 * h;                 // C-layout row
            Pld[r * 32 + ln]      = (__bf16)p0;
            Pld[r * 32 + 16 + ln] = (__bf16)p1;
        }
        __syncthreads();

        // O += P . V  (P re-read from LDS in A-fragment layout)
        v16bf ap;
        #pragma unroll
        for (int j = 0; j < 16; ++j) ap[j] = Pld[ln * 32 + a_kidx(h, j)];
        o[0] = wmma_bf16_t<false>(ap, load_bfrag(Vld, D_KEY, 0, h, ln), o[0]);
        o[1] = wmma_bf16_t<true >(ap, load_bfrag(Vld, D_KEY, 1, h, ln), o[1]);
        o[2] = wmma_bf16_t<true >(ap, load_bfrag(Vld, D_KEY, 2, h, ln), o[2]);
        o[3] = wmma_bf16_t<true >(ap, load_bfrag(Vld, D_KEY, 3, h, ln), o[3]);
        __syncthreads();
    }

    // normalize + gate-weight + scatter into dense per-expert buffer c
    #pragma unroll
    for (int i = 0; i < 8; ++i) {
        int r = i + 8 * h;
        int n = b * T_SEQ + t0 + r;
        float g = gates[n * 2 + kslot];
        int e = idxs[n * 2 + kslot];
        float inv = 1.0f / lrow[i];
        #pragma unroll
        for (int nt = 0; nt < 4; ++nt) {
            int col = nt * 16 + ln;
            c[(size_t)n * (N_EXP * D_KEY) + e * D_KEY + col] =
                (__bf16)(g * o[nt][i] * inv);
        }
    }
}

// ---------------------------------------------------------------------------
// Output projection: out[N][512] = c[N][512] @ WoFlat[512][512] + gated bias.
// (Wo[E][64][512] flattened over (e,d) is a contiguous [512][512].)
// ---------------------------------------------------------------------------
__global__ __launch_bounds__(32) void out_gemm_kernel(
    const __bf16* __restrict__ cbuf,  // [N][512]
    const __bf16* __restrict__ WoB,   // [512][512]
    const float* __restrict__ bo,     // [E][512]
    const float* __restrict__ gates, const int* __restrict__ idxs,
    float* __restrict__ out)          // [N][512]
{
    __shared__ __attribute__((aligned(32))) __bf16 Ald[16 * 32];
    __shared__ __attribute__((aligned(32))) __bf16 Bld[32 * 64];

    int lane = threadIdx.x & 31, h = lane >> 4, ln = lane & 15;
    int tile = blockIdx.x, cb = blockIdx.y;

    v8f acc[4];
    #pragma unroll
    for (int nt = 0; nt < 4; ++nt) v8f_zero(acc[nt]);

    for (int k0 = 0; k0 < D_MODEL; k0 += 32) {
#if HAS_TDM
        tdm_load_tile_bf16(Bld, WoB + (size_t)k0 * D_MODEL + cb * 64,
                           /*tensor_d0=*/D_MODEL, /*tensor_d1=*/D_MODEL - k0,
                           /*d0_stride=*/D_MODEL, /*tile_d0=*/64, /*tile_d1=*/32);
#else
        {
            const uint4* s = (const uint4*)(WoB + (size_t)(k0 + lane) * D_MODEL + cb * 64);
            uint4* d = (uint4*)(Bld + lane * 64);
            #pragma unroll
            for (int qd = 0; qd < 8; ++qd) d[qd] = s[qd];
        }
#endif
        *(v16bf*)(Ald + ln * 32 + h * 16) =
            *(const v16bf*)(cbuf + (size_t)(tile * 16 + ln) * D_MODEL + k0 + h * 16);
#if HAS_TDM
        __builtin_amdgcn_s_wait_tensorcnt(0);
#endif
        __syncthreads();

        v16bf a;
        #pragma unroll
        for (int j = 0; j < 16; ++j) a[j] = Ald[ln * 32 + a_kidx(h, j)];
        acc[0] = wmma_bf16_t<false>(a, load_bfrag(Bld, 64, 0, h, ln), acc[0]);
        acc[1] = wmma_bf16_t<true >(a, load_bfrag(Bld, 64, 1, h, ln), acc[1]);
        acc[2] = wmma_bf16_t<true >(a, load_bfrag(Bld, 64, 2, h, ln), acc[2]);
        acc[3] = wmma_bf16_t<true >(a, load_bfrag(Bld, 64, 3, h, ln), acc[3]);
        __syncthreads();
    }

    #pragma unroll
    for (int i = 0; i < 8; ++i) {
        int r = i + 8 * h;
        int n = tile * 16 + r;
        float g0 = gates[n * 2 + 0], g1 = gates[n * 2 + 1];
        int e0 = idxs[n * 2 + 0], e1 = idxs[n * 2 + 1];
        #pragma unroll
        for (int nt = 0; nt < 4; ++nt) {
            int col = cb * 64 + nt * 16 + ln;
            float bias = g0 * bo[e0 * D_MODEL + col] + g1 * bo[e1 * D_MODEL + col];
            out[(size_t)n * D_MODEL + col] = acc[nt][i] + bias;
        }
    }
}

// ---------------------------------------------------------------------------
// Host launcher
// ---------------------------------------------------------------------------
extern "C" void kernel_launch(void* const* d_in, const int* in_sizes, int n_in,
                              void* d_out, int out_size, void* d_ws, size_t ws_size,
                              hipStream_t stream) {
    (void)in_sizes; (void)n_in; (void)out_size; (void)ws_size;

    const float* q_in = (const float*)d_in[0];
    const float* k_in = (const float*)d_in[1];
    const float* v_in = (const float*)d_in[2];
    const float* Wg   = (const float*)d_in[3];
    const float* Wk   = (const float*)d_in[4];
    const float* bk   = (const float*)d_in[5];
    const float* Wv   = (const float*)d_in[6];
    const float* bv   = (const float*)d_in[7];
    const float* Wq   = (const float*)d_in[8];
    const float* bq   = (const float*)d_in[9];
    const float* Wo   = (const float*)d_in[10];
    const float* bo   = (const float*)d_in[11];
    float* out = (float*)d_out;

    char* ws = (char*)d_ws;
    size_t off = 0;
    auto alloc = [&](size_t bytes) {
        size_t o = off;
        off = (off + bytes + 255) & ~(size_t)255;
        return o;
    };
    const size_t oWk  = alloc((size_t)D_MODEL * D_KEY * 2);
    const size_t oWv  = alloc((size_t)D_MODEL * D_KEY * 2);
    const size_t oWq  = alloc((size_t)N_EXP * D_MODEL * D_KEY * 2);
    const size_t oWo  = alloc((size_t)N_EXP * D_KEY * D_MODEL * 2);
    const size_t oKh  = alloc((size_t)N_TOK * D_KEY * 2);
    const size_t oVh  = alloc((size_t)N_TOK * D_KEY * 2);
    const size_t oQa  = alloc((size_t)N_EXP * N_TOK * D_KEY * 2);
    const size_t oQh  = alloc((size_t)TOP_K * N_TOK * D_KEY * 2);
    const size_t oGat = alloc((size_t)N_TOK * TOP_K * 4);
    const size_t oIdx = alloc((size_t)N_TOK * TOP_K * 4);
    const size_t oC   = alloc((size_t)N_TOK * D_MODEL * 2);

    __bf16* WkB = (__bf16*)(ws + oWk);
    __bf16* WvB = (__bf16*)(ws + oWv);
    __bf16* WqB = (__bf16*)(ws + oWq);
    __bf16* WoB = (__bf16*)(ws + oWo);
    __bf16* kh  = (__bf16*)(ws + oKh);
    __bf16* vh  = (__bf16*)(ws + oVh);
    __bf16* qa  = (__bf16*)(ws + oQa);
    __bf16* qh  = (__bf16*)(ws + oQh);
    float*  gat = (float*)(ws + oGat);
    int*    idx = (int*)(ws + oIdx);
    __bf16* cbf = (__bf16*)(ws + oC);

    // 1) weight conversion to bf16
    {
        int n1 = D_MODEL * D_KEY;           // 32768
        int n2 = N_EXP * D_MODEL * D_KEY;   // 262144
        cvt_bf16_kernel<<<(n1 + 255) / 256, 256, 0, stream>>>(Wk, WkB, n1);
        cvt_bf16_kernel<<<(n1 + 255) / 256, 256, 0, stream>>>(Wv, WvB, n1);
        cvt_bf16_kernel<<<(n2 + 255) / 256, 256, 0, stream>>>(Wq, WqB, n2);
        cvt_bf16_kernel<<<(n2 + 255) / 256, 256, 0, stream>>>(Wo, WoB, n2);
    }
    // 2) router (softmax + top-2)
    router_kernel<<<N_TOK / 8, 256, 0, stream>>>(q_in, Wg, gat, idx);
    // 3) K/V projections + dense all-expert Q projection (WMMA + TDM)
    proj512x64_kernel<<<dim3(N_TOK / 16, 2 + N_EXP), 32, 0, stream>>>(
        q_in, k_in, v_in, WkB, WvB, WqB, bk, bv, bq, kh, vh, qa);
    // 4) gather selected expert queries (+ 1/sqrt(d) scale)
    {
        int tot = N_TOK * TOP_K * 8;
        gather_q_kernel<<<(tot + 255) / 256, 256, 0, stream>>>(qa, idx, qh);
    }
    // 5) zero dense combine buffer c
    {
        int ndw = N_TOK * D_MODEL * 2 / 4;
        zero_kernel<<<(ndw + 255) / 256, 256, 0, stream>>>((unsigned int*)cbf, ndw);
    }
    // 6) flash attention, TDM double-buffered V (WMMA)
    attn_kernel<<<dim3(T_SEQ / 16, B_SZ * TOP_K), 32, 0, stream>>>(
        qh, kh, vh, gat, idx, cbf);
    // 7) fused MoE output projection (WMMA + TDM) + gated bias
    out_gemm_kernel<<<dim3(N_TOK / 16, D_MODEL / 64), 32, 0, stream>>>(
        cbf, WoB, bo, gat, idx, out);
}
